// GNNReRanker_65197603553715
// MI455X (gfx1250) — compile-verified, matrix operand order checked by reading, not run on hardware
//
#include <hip/hip_runtime.h>
#include <hip/hip_bf16.h>
#include <math.h>

// ---------------------------------------------------------------------------
// GAT (2-layer) for MI455X / gfx1250 (wave32, WMMA).
// Layer-1 GEMM runs on v_wmma_f32_16x16x32_f16; edge phase is bandwidth-bound
// atomics. Constants match the reference.
// ---------------------------------------------------------------------------

typedef __attribute__((ext_vector_type(16))) _Float16 v16h;
typedef __attribute__((ext_vector_type(8)))  _Float16 h8;   // 16 bytes
typedef __attribute__((ext_vector_type(8)))  float    v8f;

#define D_IN      384
#define C1        256   // HEADS * HID
#define HID       128
#define NEG_SLOPE 0.2f

union AFrag { v16h v; h8 h[2]; };

// ------------------------------ utility kernels ----------------------------

__global__ void k_fill(float* __restrict__ p, long long n, float v) {
  long long i = blockIdx.x * (long long)blockDim.x + threadIdx.x;
  if (i < n) p[i] = v;
}

// f32 -> f16 convert, zero-pad rows beyond N up to Mpad
__global__ void k_cvt_x(const float* __restrict__ x, _Float16* __restrict__ xh,
                        long long nelems, long long total) {
  long long i = blockIdx.x * (long long)blockDim.x + threadIdx.x;
  if (i >= total) return;
  xh[i] = (i < nelems) ? (_Float16)x[i] : (_Float16)0.0f;
}

// W1 [384,256] row-major -> W1t_h [256,384] f16 (K-contiguous per out column)
__global__ void k_cvt_w1t(const float* __restrict__ W1, _Float16* __restrict__ w1t) {
  int i = blockIdx.x * blockDim.x + threadIdx.x;
  if (i >= D_IN * C1) return;
  int k = i >> 8;      // / 256
  int n = i & 255;     // % 256
  w1t[n * D_IN + k] = (_Float16)W1[i];
}

// ------------------------------ WMMA GEMM ----------------------------------
// C[M,256] = A[Mpad,384](f16) @ W1t^T  ; 128 threads = 4 waves per block.
// Each wave owns a 16x64 strip of one 16-row M tile: 4 accumulators,
// A-fragment reused across the 4 N-subtiles. 12 k-steps of K=32.
__global__ __launch_bounds__(128)
void k_gemm_wmma(const _Float16* __restrict__ A, const _Float16* __restrict__ Bt,
                 float* __restrict__ C, long long Mrows) {
  const int lane = threadIdx.x & 31;
  const int wave = threadIdx.x >> 5;            // 0..3 -> 64-wide N strip
  const long long mt = blockIdx.x;              // M tile
  const int r    = lane & 15;
  const int koff = (lane >> 4) << 3;            // 0 (lanes 0-15) or 8 (16-31)
  const _Float16* arow = A + (mt * 16 + r) * (long long)D_IN;
  const int n0 = wave * 64;

  v8f acc[4] = {};
  for (int k0 = 0; k0 < D_IN; k0 += 32) {
    AFrag a;
    a.h[0] = *(const h8*)(arow + k0 + koff);
    a.h[1] = *(const h8*)(arow + k0 + koff + 16);
#pragma unroll
    for (int s = 0; s < 4; ++s) {
      const _Float16* brow = Bt + (long long)(n0 + s * 16 + r) * D_IN;
      AFrag b;
      b.h[0] = *(const h8*)(brow + k0 + koff);
      b.h[1] = *(const h8*)(brow + k0 + koff + 16);
      acc[s] = __builtin_amdgcn_wmma_f32_16x16x32_f16(
          /*neg_a=*/false, a.v, /*neg_b=*/false, b.v,
          /*c_mod=*/(short)0, acc[s], /*reuse_a=*/false, /*reuse_b=*/false);
    }
  }

  // C/D layout: VGPR i -> M=i (lanes 0-15) / M=8+i (lanes 16-31); N = lane&15
  const int mo = (lane >> 4) << 3;
#pragma unroll
  for (int s = 0; s < 4; ++s) {
#pragma unroll
    for (int i = 0; i < 8; ++i) {
      long long m = mt * 16 + mo + i;
      if (m < Mrows) C[m * C1 + n0 + s * 16 + (lane & 15)] = acc[s][i];
    }
  }
}

// ------------------------- attention coefficients --------------------------

__global__ void k_alpha1(const float* __restrict__ h1, const float* __restrict__ a_src,
                         const float* __restrict__ a_dst, float* __restrict__ as1,
                         float* __restrict__ ad1, long long Nn) {
  long long t = blockIdx.x * (long long)blockDim.x + threadIdx.x;
  if (t >= Nn * 2) return;
  long long n = t >> 1;
  int hd = (int)(t & 1);
  const float* hv  = h1 + n * C1 + hd * HID;
  const float* asv = a_src + hd * HID;
  const float* adv = a_dst + hd * HID;
  float ss = 0.f, dd = 0.f;
  for (int c = 0; c < HID; ++c) { float v = hv[c]; ss += v * asv[c]; dd += v * adv[c]; }
  as1[t] = ss; ad1[t] = dd;
}

// ------------------------------ edge helpers -------------------------------

__device__ inline void edge_sd(const long long* __restrict__ ei, long long E,
                               long long e, long long& s, long long& d) {
  if (e < E) { s = ei[e]; d = ei[E + e]; } else { s = e - E; d = s; }
}

__device__ inline float lrelu(float v) { return v > 0.f ? v : NEG_SLOPE * v; }

__device__ inline void atomicMaxF(float* addr, float val) {
  int* ai = (int*)addr;
  int old = *ai;
  while (__int_as_float(old) < val) {
    int prev = atomicCAS(ai, old, __float_as_int(val));
    if (prev == old) break;
    old = prev;
  }
}

// ------------------------- layer 1 edge softmax ----------------------------

__global__ void k_edge_max1(const long long* __restrict__ ei, long long E, long long Nn,
                            const float* __restrict__ as1, const float* __restrict__ ad1,
                            float* __restrict__ m1) {
  long long e = blockIdx.x * (long long)blockDim.x + threadIdx.x;
  if (e >= E + Nn) return;
  long long s, d; edge_sd(ei, E, e, s, d);
#pragma unroll
  for (int h = 0; h < 2; ++h)
    atomicMaxF(&m1[d * 2 + h], lrelu(as1[s * 2 + h] + ad1[d * 2 + h]));
}

__global__ void k_edge_sum1(const long long* __restrict__ ei, long long E, long long Nn,
                            const float* __restrict__ as1, const float* __restrict__ ad1,
                            const float* __restrict__ m1, float* __restrict__ z1) {
  long long e = blockIdx.x * (long long)blockDim.x + threadIdx.x;
  if (e >= E + Nn) return;
  long long s, d; edge_sd(ei, E, e, s, d);
#pragma unroll
  for (int h = 0; h < 2; ++h) {
    float v = lrelu(as1[s * 2 + h] + ad1[d * 2 + h]);
    atomicAdd(&z1[d * 2 + h], __expf(v - m1[d * 2 + h]));
  }
}

// one wave per edge; lanes stride the 256 channels (coalesced)
__global__ void k_edge_agg1(const long long* __restrict__ ei, long long E, long long Nn,
                            const float* __restrict__ as1, const float* __restrict__ ad1,
                            const float* __restrict__ m1, const float* __restrict__ z1,
                            const float* __restrict__ h1, float* __restrict__ agg1) {
  long long wid = (blockIdx.x * (long long)blockDim.x + threadIdx.x) >> 5;
  int lane = threadIdx.x & 31;
  if (wid >= E + Nn) return;
  long long s, d; edge_sd(ei, E, wid, s, d);
  float w[2];
#pragma unroll
  for (int h = 0; h < 2; ++h) {
    float v = lrelu(as1[s * 2 + h] + ad1[d * 2 + h]);
    w[h] = __expf(v - m1[d * 2 + h]) / z1[d * 2 + h];
  }
  const float* hs = h1 + s * C1;
  float* od = agg1 + d * C1;
#pragma unroll
  for (int c = lane; c < C1; c += 32)
    atomicAdd(&od[c], w[c >> 7] * hs[c]);
}

// bias + ELU, in place
__global__ void k_elu(float* __restrict__ a, const float* __restrict__ b1, long long n) {
  long long i = blockIdx.x * (long long)blockDim.x + threadIdx.x;
  if (i >= n) return;
  float v = a[i] + b1[i & (C1 - 1)];
  a[i] = v > 0.f ? v : (__expf(v) - 1.f);
}

// ----------------------------- layer 2 -------------------------------------

__global__ void k_node2(const float* __restrict__ hact, const float* __restrict__ W2,
                        const float* __restrict__ a_src2, const float* __restrict__ a_dst2,
                        float* __restrict__ h2, float* __restrict__ s2,
                        float* __restrict__ d2, long long Nn) {
  long long n = blockIdx.x * (long long)blockDim.x + threadIdx.x;
  if (n >= Nn) return;
  const float* hv = hact + n * C1;
  float acc = 0.f;
  for (int c = 0; c < C1; ++c) acc += hv[c] * W2[c];
  h2[n] = acc;
  s2[n] = acc * a_src2[0];
  d2[n] = acc * a_dst2[0];
}

__global__ void k_edge_max2(const long long* __restrict__ ei, long long E, long long Nn,
                            const float* __restrict__ s2, const float* __restrict__ d2,
                            float* __restrict__ m2) {
  long long e = blockIdx.x * (long long)blockDim.x + threadIdx.x;
  if (e >= E + Nn) return;
  long long s, d; edge_sd(ei, E, e, s, d);
  atomicMaxF(&m2[d], lrelu(s2[s] + d2[d]));
}

__global__ void k_edge_sum2(const long long* __restrict__ ei, long long E, long long Nn,
                            const float* __restrict__ s2, const float* __restrict__ d2,
                            const float* __restrict__ m2, float* __restrict__ z2) {
  long long e = blockIdx.x * (long long)blockDim.x + threadIdx.x;
  if (e >= E + Nn) return;
  long long s, d; edge_sd(ei, E, e, s, d);
  atomicAdd(&z2[d], __expf(lrelu(s2[s] + d2[d]) - m2[d]));
}

__global__ void k_edge_agg2(const long long* __restrict__ ei, long long E, long long Nn,
                            const float* __restrict__ s2, const float* __restrict__ d2,
                            const float* __restrict__ m2, const float* __restrict__ z2,
                            const float* __restrict__ h2, float* __restrict__ agg2) {
  long long e = blockIdx.x * (long long)blockDim.x + threadIdx.x;
  if (e >= E + Nn) return;
  long long s, d; edge_sd(ei, E, e, s, d);
  float w = __expf(lrelu(s2[s] + d2[d]) - m2[d]) / z2[d];
  atomicAdd(&agg2[d], w * h2[s]);
}

__global__ void k_final(const float* __restrict__ agg2, const float* __restrict__ b2,
                        float* __restrict__ out, long long Nn) {
  long long n = blockIdx.x * (long long)blockDim.x + threadIdx.x;
  if (n >= Nn) return;
  out[n] = agg2[n] + b2[0];
}

// ------------------------------ launcher -----------------------------------

static inline int nblk(long long total, int bs) {
  return (int)((total + bs - 1) / bs);
}
static inline char* bump(char*& p, size_t bytes) {
  char* r = p;
  p += (bytes + 255) & ~(size_t)255;
  return r;
}

extern "C" void kernel_launch(void* const* d_in, const int* in_sizes, int n_in,
                              void* d_out, int out_size, void* d_ws, size_t ws_size,
                              hipStream_t stream) {
  const float*     x      = (const float*)d_in[0];
  const long long* ei     = (const long long*)d_in[1];   // int64 in reference
  const float*     W1     = (const float*)d_in[2];
  const float*     a_src1 = (const float*)d_in[3];
  const float*     a_dst1 = (const float*)d_in[4];
  const float*     b1     = (const float*)d_in[5];
  const float*     W2     = (const float*)d_in[6];
  const float*     a_src2 = (const float*)d_in[7];
  const float*     a_dst2 = (const float*)d_in[8];
  const float*     b2     = (const float*)d_in[9];
  float*           out    = (float*)d_out;

  const long long Nn     = (long long)in_sizes[0] / D_IN;
  const long long Ecnt   = (long long)in_sizes[1] / 2;
  const long long Mtiles = (Nn + 15) / 16;
  const long long Mpad   = Mtiles * 16;
  const long long ET     = Ecnt + Nn;   // with self-loops

  // workspace carve-out
  char* p = (char*)d_ws;
  _Float16* xh   = (_Float16*)bump(p, (size_t)Mpad * D_IN * sizeof(_Float16));
  _Float16* w1t  = (_Float16*)bump(p, (size_t)C1 * D_IN * sizeof(_Float16));
  float*    h1   = (float*)bump(p, (size_t)Mpad * C1 * sizeof(float));
  float*    as1  = (float*)bump(p, (size_t)Nn * 2 * sizeof(float));
  float*    ad1  = (float*)bump(p, (size_t)Nn * 2 * sizeof(float));
  float*    m1   = (float*)bump(p, (size_t)Nn * 2 * sizeof(float));
  float*    z1   = (float*)bump(p, (size_t)Nn * 2 * sizeof(float));
  float*    agg1 = (float*)bump(p, (size_t)Nn * C1 * sizeof(float));
  float*    h2   = (float*)bump(p, (size_t)Nn * sizeof(float));
  float*    s2   = (float*)bump(p, (size_t)Nn * sizeof(float));
  float*    d2   = (float*)bump(p, (size_t)Nn * sizeof(float));
  float*    m2   = (float*)bump(p, (size_t)Nn * sizeof(float));
  float*    z2   = (float*)bump(p, (size_t)Nn * sizeof(float));
  float*    agg2 = (float*)bump(p, (size_t)Nn * sizeof(float));

  const int BS = 256;
  const float NEG_INF = -__builtin_huge_valf();

  // init accumulators
  k_fill<<<nblk(Nn * 2, BS),  BS, 0, stream>>>(m1,   Nn * 2,  NEG_INF);
  k_fill<<<nblk(Nn * 2, BS),  BS, 0, stream>>>(z1,   Nn * 2,  0.f);
  k_fill<<<nblk(Nn * C1, BS), BS, 0, stream>>>(agg1, Nn * C1, 0.f);
  k_fill<<<nblk(Nn, BS),      BS, 0, stream>>>(m2,   Nn,      NEG_INF);
  k_fill<<<nblk(Nn, BS),      BS, 0, stream>>>(z2,   Nn,      0.f);
  k_fill<<<nblk(Nn, BS),      BS, 0, stream>>>(agg2, Nn,      0.f);

  // ---- layer 1 ----
  k_cvt_x  <<<nblk(Mpad * D_IN, BS), BS, 0, stream>>>(x, xh, Nn * D_IN, Mpad * D_IN);
  k_cvt_w1t<<<nblk(D_IN * C1, BS),   BS, 0, stream>>>(W1, w1t);
  k_gemm_wmma<<<(int)Mtiles, 128, 0, stream>>>(xh, w1t, h1, Nn);
  k_alpha1<<<nblk(Nn * 2, BS), BS, 0, stream>>>(h1, a_src1, a_dst1, as1, ad1, Nn);
  k_edge_max1<<<nblk(ET, BS), BS, 0, stream>>>(ei, Ecnt, Nn, as1, ad1, m1);
  k_edge_sum1<<<nblk(ET, BS), BS, 0, stream>>>(ei, Ecnt, Nn, as1, ad1, m1, z1);
  k_edge_agg1<<<nblk(ET * 32, BS), BS, 0, stream>>>(ei, Ecnt, Nn, as1, ad1, m1, z1, h1, agg1);
  k_elu<<<nblk(Nn * C1, BS), BS, 0, stream>>>(agg1, b1, Nn * C1);

  // ---- layer 2 ----
  k_node2<<<nblk(Nn, BS), BS, 0, stream>>>(agg1, W2, a_src2, a_dst2, h2, s2, d2, Nn);
  k_edge_max2<<<nblk(ET, BS), BS, 0, stream>>>(ei, Ecnt, Nn, s2, d2, m2);
  k_edge_sum2<<<nblk(ET, BS), BS, 0, stream>>>(ei, Ecnt, Nn, s2, d2, m2, z2);
  k_edge_agg2<<<nblk(ET, BS), BS, 0, stream>>>(ei, Ecnt, Nn, s2, d2, m2, z2, h2, agg2);
  k_final<<<nblk(Nn, BS), BS, 0, stream>>>(agg2, b2, out, Nn);
}
